// LSTMModel_32238024524521
// MI455X (gfx1250) — compile-verified, matrix operand order checked by reading, not run on hardware
//
#include <hip/hip_runtime.h>
#include <hip/hip_bf16.h>

// ---------------------------------------------------------------------------
// Problem constants (match reference)
// ---------------------------------------------------------------------------
#define TT 256
#define BB 64
#define II 512
#define HH 1024
#define OO 512
#define H4 (4 * HH)

typedef __bf16 bf16;
typedef bf16  v16bf __attribute__((ext_vector_type(16)));
typedef bf16  v8bf  __attribute__((ext_vector_type(8)));
typedef float v8f   __attribute__((ext_vector_type(8)));
typedef int   v4i   __attribute__((vector_size(16)));   // matches async-LDS builtin param

// gfx1250 async global->LDS path (guarded; falls back to plain copy)
#if defined(__has_builtin)
#  if __has_builtin(__builtin_amdgcn_global_load_async_to_lds_b128) && \
      __has_builtin(__builtin_amdgcn_s_wait_asynccnt)
#    define USE_ASYNC_LDS 1
#  endif
#endif
#ifndef USE_ASYNC_LDS
#  define USE_ASYNC_LDS 0
#endif

// ---------------------------------------------------------------------------
// WMMA bf16 fragment loader (works for global or LDS-backed pointers; the
// address space is inferred per-callsite after inlining).
// A (16x32, 16-bit): lane L holds row (L&15); k-chunks [8*(L>>4), +8) and
// [8*(L>>4)+16, +8). A row-major [N x K] weight matrix is directly the B
// operand (column n of B == row n of W) with the identical per-lane pattern.
// ---------------------------------------------------------------------------
__device__ __forceinline__ v16bf load_frag(const bf16* __restrict__ tile, int ld, int lane) {
    const bf16* p = tile + (size_t)(lane & 15) * ld + ((lane >> 4) << 3);
    v8bf lo = *reinterpret_cast<const v8bf*>(p);
    v8bf hi = *reinterpret_cast<const v8bf*>(p + 16);
    return __builtin_shufflevector(lo, hi, 0, 1, 2, 3, 4, 5, 6, 7,
                                          8, 9, 10, 11, 12, 13, 14, 15);
}

// ---------------------------------------------------------------------------
// Generic GEMM: C[M,N] = A[M,K](bf16) * W[N,K]^T(bf16) + bias[N], fp32 out.
// Block = 256 threads = 8 waves, block tile 128x128, wave tile 64x32.
// ---------------------------------------------------------------------------
__global__ void __launch_bounds__(256)
gemm_bf16_wmma(const bf16* __restrict__ A, const bf16* __restrict__ W,
               const float* __restrict__ bias, float* __restrict__ C,
               int M, int N, int K) {
    const int lane = threadIdx.x & 31;
    const int wid  = threadIdx.x >> 5;
    const int m0 = blockIdx.y * 128 + (wid >> 2) * 64;
    const int n0 = blockIdx.x * 128 + (wid & 3) * 32;

    v8f acc[4][2] = {};
    for (int k0 = 0; k0 < K; k0 += 32) {
        v16bf a[4], b[2];
#pragma unroll
        for (int i = 0; i < 4; ++i)
            a[i] = load_frag(A + (size_t)(m0 + 16 * i) * K + k0, K, lane);
#pragma unroll
        for (int j = 0; j < 2; ++j)
            b[j] = load_frag(W + (size_t)(n0 + 16 * j) * K + k0, K, lane);
#pragma unroll
        for (int i = 0; i < 4; ++i)
#pragma unroll
            for (int j = 0; j < 2; ++j)
                acc[i][j] = __builtin_amdgcn_wmma_f32_16x16x32_bf16(
                    false, a[i], false, b[j], (short)0, acc[i][j], false, false);
    }

    const int nn = lane & 15;
    const int mo = 8 * (lane >> 4);
#pragma unroll
    for (int i = 0; i < 4; ++i) {
#pragma unroll
        for (int j = 0; j < 2; ++j) {
            const int nc = n0 + 16 * j + nn;
            const float bv = bias[nc];
#pragma unroll
            for (int r = 0; r < 8; ++r) {
                const int mc = m0 + 16 * i + mo + r;
                C[(size_t)mc * N + nc] = acc[i][j][r] + bv;
            }
        }
    }
}

// ---------------------------------------------------------------------------
// Persistent LSTM layer kernel: runs ALL T timesteps in one launch.
//
//   Grid: 64 blocks x 128 threads (4 waves). Block b owns hidden slice
//   j0 = b*16; wave w owns batch tile mt = w*16 and computes all 4 gate
//   tiles for (mt, j0) so the nonlinearity fuses in-register.
//
//   The block's full weight working set w_hh[{i,f,g,o} x 16 rows x K=1024]
//   (128 KB bf16) is staged into LDS ONCE (async global->LDS when
//   available), then every timestep reads B-fragments via ds_load_b128.
//   Global weight traffic for the whole recurrence: 16 MB instead of ~4 GB.
//
//   Steps are separated by a software grid barrier (release fence ->
//   atomic arrive -> spin -> acquire fence) so h[t] written by all blocks
//   is visible before step t+1. 64 blocks @128 KB LDS are trivially
//   co-resident on a 256-WGP part (320 KB LDS per WGP).
// ---------------------------------------------------------------------------
__global__ void __launch_bounds__(128)
lstm_layer_persistent(const float* __restrict__ xp,    // [T, B, 4H] (biases folded in)
                      const bf16*  __restrict__ whh,   // [4H, H] bf16 row-major
                      bf16*        __restrict__ hseq,  // [T+1, B, H]; row 0 zeroed
                      float*       __restrict__ c,     // [B, H] zeroed
                      int*         __restrict__ ctr) { // grid barrier counter (zeroed)
    extern __shared__ char smem[];
    bf16* w_lds = (bf16*)smem;                         // [4][16][1024]

    const int lane = threadIdx.x & 31;
    const int wid  = threadIdx.x >> 5;
    const int j0   = blockIdx.x * 16;                  // hidden slice origin
    const int mt   = wid * 16;                         // batch tile origin
    const int nblk = gridDim.x;

    // ---- stage w_hh slice into LDS (64 rows x 2 KB = 8192 16-byte chunks) --
    for (int i = threadIdx.x; i < 8192; i += 128) {
        const int r64 = i >> 7;                        // 0..63 : g*16 + r
        const int off = (i & 127) << 4;                // byte offset within row
        const char* g = (const char*)(whh +
            (size_t)((r64 >> 4) * HH + j0 + (r64 & 15)) * HH) + off;
        char* l = (char*)w_lds + ((size_t)r64 << 11) + off;
#if USE_ASYNC_LDS
        __builtin_amdgcn_global_load_async_to_lds_b128((v4i*)g, (v4i*)l, 0, 0);
#else
        *(v4i*)l = *(const v4i*)g;
#endif
    }
#if USE_ASYNC_LDS
    __builtin_amdgcn_s_wait_asynccnt(0);
#endif
    __syncthreads();

    const int n    = j0 + (lane & 15);
    const int mrow = mt + 8 * (lane >> 4);
    const size_t BH = (size_t)BB * HH;

    for (int t = 0; t < TT; ++t) {
        const float* xpt    = xp + (size_t)t * BB * H4;
        const bf16*  h_prev = hseq + (size_t)t * BH;
        bf16*        h_out  = hseq + (size_t)(t + 1) * BH;

        // prefetch next timestep's x_proj slice (1 MB across 8192 lanes)
        if (t + 1 < TT) {
            const float* nxt = xp + (size_t)(t + 1) * BB * H4;
            __builtin_prefetch(nxt + ((size_t)(blockIdx.x * 128 + threadIdx.x) << 5), 0, 0);
        }

        // ---- gates[mt:mt+16, {i,f,g,o} @ j0] = h_prev * w_hh^T -------------
        v8f acc[4] = {};
        for (int k0 = 0; k0 < HH; k0 += 32) {
            const v16bf a = load_frag(h_prev + (size_t)mt * HH + k0, HH, lane);
#pragma unroll
            for (int g = 0; g < 4; ++g) {
                const v16bf b = load_frag(w_lds + (g << 14) + k0, HH, lane); // ds_load
                acc[g] = __builtin_amdgcn_wmma_f32_16x16x32_bf16(
                    false, a, false, b, (short)0, acc[g], false, false);
            }
        }

        // ---- fused gate nonlinearity + state update ------------------------
#pragma unroll
        for (int r = 0; r < 8; ++r) {
            const int m = mrow + r;
            float gi = acc[0][r] + xpt[(size_t)m * H4 + 0 * HH + n];
            float gf = acc[1][r] + xpt[(size_t)m * H4 + 1 * HH + n];
            float gg = acc[2][r] + xpt[(size_t)m * H4 + 2 * HH + n];
            float go = acc[3][r] + xpt[(size_t)m * H4 + 3 * HH + n];
            gi = 1.0f / (1.0f + __expf(-gi));
            gf = 1.0f / (1.0f + __expf(-gf));
            gg = tanhf(gg);
            go = 1.0f / (1.0f + __expf(-go));
            const size_t idx = (size_t)m * HH + n;
            const float cc = gf * c[idx] + gi * gg;
            c[idx] = cc;
            h_out[idx] = (bf16)(go * tanhf(cc));
        }

        // ---- grid-wide barrier: make h[t+1] visible to all blocks ----------
        __syncthreads();
        if (threadIdx.x == 0) {
            __builtin_amdgcn_fence(__ATOMIC_RELEASE, "agent");
            __hip_atomic_fetch_add(ctr, 1, __ATOMIC_RELAXED, __HIP_MEMORY_SCOPE_AGENT);
            const int target = nblk * (t + 1);
            while (__hip_atomic_load(ctr, __ATOMIC_RELAXED, __HIP_MEMORY_SCOPE_AGENT) < target)
                __builtin_amdgcn_s_sleep(2);
            __builtin_amdgcn_fence(__ATOMIC_ACQUIRE, "agent");
        }
        __syncthreads();
    }
}

// ---------------------------------------------------------------------------
// Small helper kernels
// ---------------------------------------------------------------------------
__global__ void cvt_f32_to_bf16(const float* __restrict__ src, bf16* __restrict__ dst, int n) {
    const int i = blockIdx.x * blockDim.x + threadIdx.x;
    if (i < n) dst[i] = (bf16)src[i];
}

__global__ void combine_bias(const float* __restrict__ a, const float* __restrict__ b,
                             float* __restrict__ out, int n) {
    const int i = blockIdx.x * blockDim.x + threadIdx.x;
    if (i < n) out[i] = a[i] + b[i];
}

__global__ void zero_bf16(bf16* __restrict__ p, int n) {
    const int i = blockIdx.x * blockDim.x + threadIdx.x;
    if (i < n) p[i] = (bf16)0.0f;
}

__global__ void zero_f32(float* __restrict__ p, int n) {
    const int i = blockIdx.x * blockDim.x + threadIdx.x;
    if (i < n) p[i] = 0.0f;
}

// ---------------------------------------------------------------------------
// Orchestration
// ---------------------------------------------------------------------------
extern "C" void kernel_launch(void* const* d_in, const int* in_sizes, int n_in,
                              void* d_out, int out_size, void* d_ws, size_t ws_size,
                              hipStream_t stream) {
    const float* data  = (const float*)d_in[0];   // [T,B,I]
    const float* w_ih0 = (const float*)d_in[1];   // [4H,I]
    const float* w_hh0 = (const float*)d_in[2];   // [4H,H]
    const float* b_ih0 = (const float*)d_in[3];
    const float* b_hh0 = (const float*)d_in[4];
    const float* w_ih1 = (const float*)d_in[5];   // [4H,H]
    const float* w_hh1 = (const float*)d_in[6];   // [4H,H]
    const float* b_ih1 = (const float*)d_in[7];
    const float* b_hh1 = (const float*)d_in[8];
    const float* w_out = (const float*)d_in[9];   // [O,H]
    const float* b_out = (const float*)d_in[10];
    float* out = (float*)d_out;                   // [T,B,O]

    // --- workspace carve-up -------------------------------------------------
    char*  ws  = (char*)d_ws;
    size_t off = 0;
    auto alloc = [&](size_t bytes) -> void* {
        off = (off + 255) & ~(size_t)255;
        void* p = ws + off;
        off += bytes;
        return p;
    };
    const size_t MB = (size_t)TT * BB;            // 16384 rows
    bf16*  data_bf = (bf16*)alloc(MB * II * sizeof(bf16));
    bf16*  wih0_bf = (bf16*)alloc((size_t)H4 * II * sizeof(bf16));
    bf16*  whh0_bf = (bf16*)alloc((size_t)H4 * HH * sizeof(bf16));
    bf16*  wih1_bf = (bf16*)alloc((size_t)H4 * HH * sizeof(bf16));
    bf16*  whh1_bf = (bf16*)alloc((size_t)H4 * HH * sizeof(bf16));
    bf16*  wout_bf = (bf16*)alloc((size_t)OO * HH * sizeof(bf16));
    float* bias0   = (float*)alloc((size_t)H4 * sizeof(float));
    float* bias1   = (float*)alloc((size_t)H4 * sizeof(float));
    float* xp      = (float*)alloc(MB * H4 * sizeof(float));          // reused L0/L1
    bf16*  h1seq   = (bf16*)alloc((size_t)(TT + 1) * BB * HH * sizeof(bf16));
    bf16*  h2seq   = (bf16*)alloc((size_t)(TT + 1) * BB * HH * sizeof(bf16));
    float* cstate  = (float*)alloc((size_t)BB * HH * sizeof(float));
    int*   barrier = (int*)alloc(256);

    const int TH = 256;
    auto blocks1d = [](size_t n, int t) { return (unsigned)((n + t - 1) / t); };

    // allow 128 KB dynamic LDS on the persistent kernel (host-side, idempotent)
    (void)hipFuncSetAttribute((const void*)lstm_layer_persistent,
                              hipFuncAttributeMaxDynamicSharedMemorySize, 131072);

    // --- precision conversion + bias folding -------------------------------
    cvt_f32_to_bf16<<<blocks1d(MB * II, TH), TH, 0, stream>>>(data, data_bf, (int)(MB * II));
    cvt_f32_to_bf16<<<blocks1d((size_t)H4 * II, TH), TH, 0, stream>>>(w_ih0, wih0_bf, H4 * II);
    cvt_f32_to_bf16<<<blocks1d((size_t)H4 * HH, TH), TH, 0, stream>>>(w_hh0, whh0_bf, H4 * HH);
    cvt_f32_to_bf16<<<blocks1d((size_t)H4 * HH, TH), TH, 0, stream>>>(w_ih1, wih1_bf, H4 * HH);
    cvt_f32_to_bf16<<<blocks1d((size_t)H4 * HH, TH), TH, 0, stream>>>(w_hh1, whh1_bf, H4 * HH);
    cvt_f32_to_bf16<<<blocks1d((size_t)OO * HH, TH), TH, 0, stream>>>(w_out, wout_bf, OO * HH);
    combine_bias<<<blocks1d(H4, TH), TH, 0, stream>>>(b_ih0, b_hh0, bias0, H4);
    combine_bias<<<blocks1d(H4, TH), TH, 0, stream>>>(b_ih1, b_hh1, bias1, H4);
    zero_bf16<<<blocks1d((size_t)BB * HH, TH), TH, 0, stream>>>(h1seq, BB * HH);  // h[-1]=0
    zero_bf16<<<blocks1d((size_t)BB * HH, TH), TH, 0, stream>>>(h2seq, BB * HH);

    const size_t BH = (size_t)BB * HH;

    // --- layer 0: input projection GEMM, then persistent recurrence --------
    {
        dim3 grid(H4 / 128, (unsigned)(MB / 128));
        gemm_bf16_wmma<<<grid, 256, 0, stream>>>(data_bf, wih0_bf, bias0, xp,
                                                 (int)MB, H4, II);
    }
    zero_f32<<<blocks1d(BH, TH), TH, 0, stream>>>(cstate, (int)BH);
    zero_f32<<<1, 1, 0, stream>>>((float*)barrier, 1);
    lstm_layer_persistent<<<64, 128, 131072, stream>>>(xp, whh0_bf, h1seq, cstate, barrier);

    // --- layer 1 ------------------------------------------------------------
    {
        dim3 grid(H4 / 128, (unsigned)(MB / 128));
        gemm_bf16_wmma<<<grid, 256, 0, stream>>>(h1seq + BH, wih1_bf, bias1, xp,
                                                 (int)MB, H4, HH);
    }
    zero_f32<<<blocks1d(BH, TH), TH, 0, stream>>>(cstate, (int)BH);
    zero_f32<<<1, 1, 0, stream>>>((float*)barrier, 1);
    lstm_layer_persistent<<<64, 128, 131072, stream>>>(xp, whh1_bf, h2seq, cstate, barrier);

    // --- output projection --------------------------------------------------
    {
        dim3 grid(OO / 128, (unsigned)(MB / 128));
        gemm_bf16_wmma<<<grid, 256, 0, stream>>>(h2seq + BH, wout_bf, b_out, out,
                                                 (int)MB, OO, HH);
    }
}
